// TransformerVAE_37194416783412
// MI455X (gfx1250) — compile-verified, matrix operand order checked by reading, not run on hardware
//
#include <hip/hip_runtime.h>
#include <hip/hip_bf16.h>

typedef __attribute__((ext_vector_type(16))) _Float16 v16h;
typedef __attribute__((ext_vector_type(8)))  float    v8f;

#define D_MODEL 512
#define SEQ     128
#define BATCH   512
#define NTOK    (BATCH * SEQ)      // 65536 tokens
#define D_FF    2048
#define LATENT  256
#define NHEAD   8
#define DK      64
#define NCAT    32
#define CATD    16
#define FFCHUNK 8192

// ---------------------------------------------------------------------------
// WMMA helpers (CDNA5 wave32, V_WMMA_F32_16X16X32_F16)
// A-fragment layout (ISA 7.12.2, 16-bit A 16x32): lane L holds row M=L&15,
// K-half = (L>>4)*8; element e -> K = (e<8?0:16) + khalf + (e&7).
// B staged transposed in LDS ([n][k]) so B fragments load identically.
// C/D layout: VGPR r -> row r + (L>>4)*8, lane&15 -> column.
// ---------------------------------------------------------------------------
__device__ inline v16h frag_ld(const _Float16* __restrict__ row, int khalf) {
  v16h f;
#pragma unroll
  for (int e = 0; e < 16; ++e) {
    int kk = ((e < 8) ? 0 : 16) + khalf + (e & 7);
    f[e] = row[kk];
  }
  return f;
}

__device__ inline v8f wmma16(v16h a, v16h b, v8f c) {
  return __builtin_amdgcn_wmma_f32_16x16x32_f16(false, a, false, b,
                                                (short)0, c, false, false);
}

__device__ inline v8f v8f_zero() {
  v8f z = {0.f, 0.f, 0.f, 0.f, 0.f, 0.f, 0.f, 0.f};
  return z;
}

// ---------------------------------------------------------------------------
// Generic GEMM: C[M,N] = A[M,K] @ W[K,N] + bias[N], optional ReLU.
// Block tile 128x128, 8 waves, each wave computes a 32x64 subtile (2x4 WMMA
// per 32-deep k-step). Double-buffered LDS + software pipeline: tile t+1's
// global loads are issued before tile t's WMMAs; converts/stores to the
// alternate LDS buffer land while the matrix pipe is busy.
// All M%128==0, N%128==0, K%32==0 for every call in this network.
// ---------------------------------------------------------------------------
__global__ __launch_bounds__(256) void gemm_wmma_kernel(
    const float* __restrict__ A, const float* __restrict__ W,
    const float* __restrict__ bias, float* __restrict__ C,
    int M, int N, int K, int relu)
{
  __shared__ _Float16 sA[2][128][34];   // [m][k], +2 pad
  __shared__ _Float16 sB[2][128][34];   // transposed: [n][k]

  const int tid    = threadIdx.x;
  const int lane   = tid & 31;
  const int wave   = tid >> 5;
  const int wm     = wave & 3;          // 4 M-subtiles of 32 rows
  const int wn     = wave >> 2;         // 2 N-subtiles of 64 cols
  const int lane16 = lane & 15;
  const int khalf  = (lane >> 4) * 8;
  const int bm     = blockIdx.y * 128;
  const int bn     = blockIdx.x * 128;

  float4 Areg[4], Breg[4];

  // Issue all global loads for one k-tile (no waits interleaved).
  auto ld_tiles = [&](int k0) {
#pragma unroll
    for (int j = 0; j < 4; ++j) {
      int c = tid + j * 256;                         // 1024 float4 chunks each
      Areg[j] = *(const float4*)(A + (size_t)(bm + (c >> 3)) * K + k0 + ((c & 7) << 2));
      Breg[j] = *(const float4*)(W + (size_t)(k0 + (c >> 5)) * N + bn + ((c & 31) << 2));
    }
  };
  // Convert fp32 -> f16 and store to LDS buffer `buf`.
  auto st_tiles = [&](int buf) {
#pragma unroll
    for (int j = 0; j < 4; ++j) {
      int c  = tid + j * 256;
      int ar = c >> 3, ak = (c & 7) << 2;            // A: row 0..127, k 0..28
      sA[buf][ar][ak + 0] = (_Float16)Areg[j].x;
      sA[buf][ar][ak + 1] = (_Float16)Areg[j].y;
      sA[buf][ar][ak + 2] = (_Float16)Areg[j].z;
      sA[buf][ar][ak + 3] = (_Float16)Areg[j].w;
      int bk = c >> 5, bn0 = (c & 31) << 2;          // B: k 0..31, n 0..124
      sB[buf][bn0 + 0][bk] = (_Float16)Breg[j].x;
      sB[buf][bn0 + 1][bk] = (_Float16)Breg[j].y;
      sB[buf][bn0 + 2][bk] = (_Float16)Breg[j].z;
      sB[buf][bn0 + 3][bk] = (_Float16)Breg[j].w;
    }
  };

  v8f acc[2][4];
#pragma unroll
  for (int mi = 0; mi < 2; ++mi)
#pragma unroll
    for (int nj = 0; nj < 4; ++nj) acc[mi][nj] = v8f_zero();

  const int nk = K >> 5;
  ld_tiles(0);
  st_tiles(0);
  __syncthreads();

  for (int t = 0; t < nk; ++t) {
    const int cur = t & 1;
    if (t + 1 < nk) ld_tiles((t + 1) << 5);          // overlap with WMMAs below
    if (t + 2 < nk) {                                // gfx1250 global_prefetch_b8
      __builtin_prefetch(A + (size_t)(bm + (tid >> 1)) * K + ((t + 2) << 5), 0, 1);
      __builtin_prefetch(W + (size_t)(((t + 2) << 5) + (tid >> 3)) * N + bn + ((tid & 7) << 4), 0, 1);
    }

    v16h a0 = frag_ld(&sA[cur][wm * 32 + lane16][0], khalf);
    v16h a1 = frag_ld(&sA[cur][wm * 32 + 16 + lane16][0], khalf);
#pragma unroll
    for (int nj = 0; nj < 4; ++nj) {
      v16h bf = frag_ld(&sB[cur][wn * 64 + nj * 16 + lane16][0], khalf);
      acc[0][nj] = wmma16(a0, bf, acc[0][nj]);
      acc[1][nj] = wmma16(a1, bf, acc[1][nj]);
    }

    if (t + 1 < nk) st_tiles(cur ^ 1);               // write the other buffer
    __syncthreads();
  }

  // Epilogue: bias (+ ReLU), fp32 store
#pragma unroll
  for (int mi = 0; mi < 2; ++mi)
#pragma unroll
    for (int nj = 0; nj < 4; ++nj)
#pragma unroll
      for (int r = 0; r < 8; ++r) {
        int row = bm + wm * 32 + mi * 16 + khalf + r;
        int col = bn + wn * 64 + nj * 16 + lane16;
        float v = acc[mi][nj][r] + bias[col];
        if (relu) v = fmaxf(v, 0.f);
        C[(size_t)row * N + col] = v;
      }
}

// ---------------------------------------------------------------------------
// Fused attention, one block per (batch, head). S=128, Dk=64.
// scores = (Q*0.125) Kt via WMMA, register softmax, ctx = P V via WMMA.
// V is loaded into registers up front so its global latency hides behind the
// score WMMAs and softmax; it is committed to LDS only in phase 2.
// ---------------------------------------------------------------------------
struct AttnP1 { _Float16 q[128][68]; _Float16 k[128][68]; };   // 34.0 KB
struct AttnP2 { _Float16 p[128][132]; _Float16 vt[64][132]; }; // 50.7 KB
union AttnSmem { AttnP1 p1; AttnP2 p2; };

__global__ __launch_bounds__(256) void attn_kernel(
    const float* __restrict__ Q, const float* __restrict__ Kg,
    const float* __restrict__ V, float* __restrict__ O)
{
  __shared__ AttnSmem sm;
  const int tid    = threadIdx.x;
  const int lane   = tid & 31;
  const int wave   = tid >> 5;          // 8 waves -> 16 query rows each
  const int lane16 = lane & 15;
  const int khalf  = (lane >> 4) * 8;
  const int b      = blockIdx.x >> 3;
  const int hd     = blockIdx.x & 7;
  const size_t base = ((size_t)b * SEQ) * D_MODEL + (size_t)hd * DK;

  // Phase 1: issue Q/K/V loads (float4, 2048 chunks each), then store Q,K.
  float4 qr[8], kr[8], vr[8];
#pragma unroll
  for (int j = 0; j < 8; ++j) {
    int c = tid + j * 256;
    int s = c >> 4;                     // 16 float4 per 64-wide row
    int d = (c & 15) << 2;
    const size_t off = base + (size_t)s * D_MODEL + d;
    qr[j] = *(const float4*)(Q + off);
    kr[j] = *(const float4*)(Kg + off);
    vr[j] = *(const float4*)(V + off);
  }
#pragma unroll
  for (int j = 0; j < 8; ++j) {
    int c = tid + j * 256;
    int s = c >> 4;
    int d = (c & 15) << 2;
    sm.p1.q[s][d + 0] = (_Float16)(0.125f * qr[j].x);
    sm.p1.q[s][d + 1] = (_Float16)(0.125f * qr[j].y);
    sm.p1.q[s][d + 2] = (_Float16)(0.125f * qr[j].z);
    sm.p1.q[s][d + 3] = (_Float16)(0.125f * qr[j].w);
    sm.p1.k[s][d + 0] = (_Float16)kr[j].x;
    sm.p1.k[s][d + 1] = (_Float16)kr[j].y;
    sm.p1.k[s][d + 2] = (_Float16)kr[j].z;
    sm.p1.k[s][d + 3] = (_Float16)kr[j].w;
  }
  __syncthreads();

  v16h qf0 = frag_ld(&sm.p1.q[wave * 16 + lane16][0],  khalf);
  v16h qf1 = frag_ld(&sm.p1.q[wave * 16 + lane16][32], khalf);

  v8f sc[8];
#pragma unroll
  for (int j = 0; j < 8; ++j) sc[j] = v8f_zero();
#pragma unroll
  for (int j = 0; j < 8; ++j) {         // 8 key subtiles of 16
    v16h k0 = frag_ld(&sm.p1.k[j * 16 + lane16][0],  khalf);
    v16h k1 = frag_ld(&sm.p1.k[j * 16 + lane16][32], khalf);
    sc[j] = wmma16(qf0, k0, sc[j]);
    sc[j] = wmma16(qf1, k1, sc[j]);
  }

  // Softmax over 128 columns. Row values for accumulator element r live
  // across fragments j and the 16 lanes of this half-wave.
  float mx[8], sum[8], rinv[8];
#pragma unroll
  for (int r = 0; r < 8; ++r) mx[r] = -3.0e38f;
#pragma unroll
  for (int j = 0; j < 8; ++j)
#pragma unroll
    for (int r = 0; r < 8; ++r) mx[r] = fmaxf(mx[r], sc[j][r]);
#pragma unroll
  for (int off = 1; off < 16; off <<= 1)
#pragma unroll
    for (int r = 0; r < 8; ++r) mx[r] = fmaxf(mx[r], __shfl_xor(mx[r], off, 32));
#pragma unroll
  for (int r = 0; r < 8; ++r) sum[r] = 0.f;
#pragma unroll
  for (int j = 0; j < 8; ++j)
#pragma unroll
    for (int r = 0; r < 8; ++r) {
      float e = __expf(sc[j][r] - mx[r]);
      sc[j][r] = e;
      sum[r] += e;
    }
#pragma unroll
  for (int off = 1; off < 16; off <<= 1)
#pragma unroll
    for (int r = 0; r < 8; ++r) sum[r] += __shfl_xor(sum[r], off, 32);
#pragma unroll
  for (int r = 0; r < 8; ++r) rinv[r] = 1.f / sum[r];

  __syncthreads();   // everyone done reading p1 before overwriting as p2

  // Phase 2: write normalized P (f16) and commit V transposed from registers
#pragma unroll
  for (int j = 0; j < 8; ++j)
#pragma unroll
    for (int r = 0; r < 8; ++r) {
      int row = wave * 16 + khalf + r;
      int col = j * 16 + lane16;
      sm.p2.p[row][col] = (_Float16)(sc[j][r] * rinv[r]);
    }
#pragma unroll
  for (int j = 0; j < 8; ++j) {
    int c = tid + j * 256;
    int s = c >> 4;
    int d = (c & 15) << 2;
    sm.p2.vt[d + 0][s] = (_Float16)vr[j].x;
    sm.p2.vt[d + 1][s] = (_Float16)vr[j].y;
    sm.p2.vt[d + 2][s] = (_Float16)vr[j].z;
    sm.p2.vt[d + 3][s] = (_Float16)vr[j].w;
  }
  __syncthreads();

  v8f o[4];
#pragma unroll
  for (int n = 0; n < 4; ++n) o[n] = v8f_zero();
#pragma unroll
  for (int kk = 0; kk < 4; ++kk) {      // K=128 in 4 steps of 32
    v16h af = frag_ld(&sm.p2.p[wave * 16 + lane16][kk * 32], khalf);
#pragma unroll
    for (int n = 0; n < 4; ++n) {
      v16h bf = frag_ld(&sm.p2.vt[n * 16 + lane16][kk * 32], khalf);
      o[n] = wmma16(af, bf, o[n]);
    }
  }
#pragma unroll
  for (int n = 0; n < 4; ++n)
#pragma unroll
    for (int r = 0; r < 8; ++r) {
      int s = wave * 16 + khalf + r;
      int d = n * 16 + lane16;
      O[base + (size_t)s * D_MODEL + d] = o[n][r];
    }
}

// ---------------------------------------------------------------------------
// Elementwise / reduction kernels
// ---------------------------------------------------------------------------
__global__ __launch_bounds__(128) void embed_kernel(
    const float* __restrict__ x, const float* __restrict__ fw,
    const float* __restrict__ fb, const float* __restrict__ pos,
    float* __restrict__ H)
{
  int row = blockIdx.x;            // token = b*128 + s
  int s   = row & (SEQ - 1);
  float xv = x[row];
  int t = threadIdx.x;
#pragma unroll
  for (int i = 0; i < 4; ++i) {
    int d = t + i * 128;
    H[(size_t)row * D_MODEL + d] = xv * fw[d] + fb[d] + pos[(size_t)s * D_MODEL + d];
  }
}

// h = LN(h + res) * g + be, in place on h. One block per token row.
__global__ __launch_bounds__(128) void ln_residual_kernel(
    float* __restrict__ h, const float* __restrict__ res,
    const float* __restrict__ g, const float* __restrict__ be)
{
  __shared__ float rs[4], rs2[4];
  const size_t row = blockIdx.x;
  const int t = threadIdx.x;
  float v[4];
  float s = 0.f, s2 = 0.f;
#pragma unroll
  for (int i = 0; i < 4; ++i) {
    int d = t + i * 128;
    float a = h[row * D_MODEL + d] + res[row * D_MODEL + d];
    v[i] = a; s += a; s2 += a * a;
  }
#pragma unroll
  for (int off = 1; off < 32; off <<= 1) {
    s  += __shfl_xor(s,  off, 32);
    s2 += __shfl_xor(s2, off, 32);
  }
  if ((t & 31) == 0) { rs[t >> 5] = s; rs2[t >> 5] = s2; }
  __syncthreads();
  float S  = rs[0] + rs[1] + rs[2] + rs[3];
  float S2 = rs2[0] + rs2[1] + rs2[2] + rs2[3];
  float m    = S * (1.f / D_MODEL);
  float var  = S2 * (1.f / D_MODEL) - m * m;
  float rstd = rsqrtf(var + 1e-5f);
#pragma unroll
  for (int i = 0; i < 4; ++i) {
    int d = t + i * 128;
    h[row * D_MODEL + d] = (v[i] - m) * rstd * g[d] + be[d];
  }
}

__global__ __launch_bounds__(256) void reparam_kernel(
    const float* __restrict__ mu, const float* __restrict__ lv,
    const float* __restrict__ eps, float* __restrict__ z)
{
  int i = blockIdx.x * 256 + threadIdx.x;
  z[i] = mu[i] + eps[i] * __expf(0.5f * lv[i]);
}

// recon[b, num_idx[j]] = dot(d[b, num_idx[j], :], num_W) + num_b
__global__ __launch_bounds__(128) void num_head_kernel(
    const float* __restrict__ H, const float* __restrict__ numW,
    const float* __restrict__ numB, const int* __restrict__ num_idx,
    float* __restrict__ recon)
{
  __shared__ float red[4];
  int b = blockIdx.x / 96;
  int j = blockIdx.x % 96;
  int token = num_idx[j];
  const float* row = H + ((size_t)b * SEQ + token) * D_MODEL;
  int t = threadIdx.x;
  float s = 0.f;
#pragma unroll
  for (int i = 0; i < 4; ++i) {
    int d = t + i * 128;
    s += row[d] * numW[d];
  }
#pragma unroll
  for (int off = 1; off < 32; off <<= 1) s += __shfl_xor(s, off, 32);
  if ((t & 31) == 0) red[t >> 5] = s;
  __syncthreads();
  if (t == 0)
    recon[(size_t)b * SEQ + token] = red[0] + red[1] + red[2] + red[3] + numB[0];
}

// Gumbel-softmax categorical head; one block per (b, n).
__global__ __launch_bounds__(256) void cat_head_kernel(
    const float* __restrict__ H, const float* __restrict__ catW,
    const float* __restrict__ catB, const float* __restrict__ gum,
    const int* __restrict__ cat_idx, float* __restrict__ recon)
{
  __shared__ float sl[16];
  const int tid = threadIdx.x;
  const int b = blockIdx.x >> 5;
  const int n = blockIdx.x & 31;
  const int k = tid >> 4;       // 0..15 class
  const int sub = tid & 15;     // 16 partial lanes per class
  const int token = cat_idx[n];
  const float* row = H + ((size_t)b * SEQ + token) * D_MODEL;
  float p = 0.f;
  for (int i = 0; i < 32; ++i) {
    int d = sub * 32 + i;
    p += row[d] * catW[((size_t)n * D_MODEL + d) * CATD + k];
  }
#pragma unroll
  for (int off = 1; off < 16; off <<= 1) p += __shfl_xor(p, off, 32);
  if (sub == 0) sl[k] = p + catB[n * CATD + k];
  __syncthreads();
  if (tid < 16) {
    float l = sl[tid] + gum[((size_t)b * NCAT + n) * CATD + tid];   // TAU == 1
    float m = l;
#pragma unroll
    for (int off = 1; off < 16; off <<= 1) m = fmaxf(m, __shfl_xor(m, off, 32));
    float e = __expf(l - m);
    float s = e;
#pragma unroll
    for (int off = 1; off < 16; off <<= 1) s += __shfl_xor(s, off, 32);
    float cls = (e / s) * (float)tid;
#pragma unroll
    for (int off = 1; off < 16; off <<= 1) cls += __shfl_xor(cls, off, 32);
    if (tid == 0) recon[(size_t)b * SEQ + token] = cls;
  }
}

// ---------------------------------------------------------------------------
// Host orchestration
// ---------------------------------------------------------------------------
struct Blk { const float* p[16]; };
// leaf order: Wq,bq,Wk,bk,Wv,bv,Wo,bo,g1,be1,g2,be2,W1,bf1,W2,bf2

extern "C" void kernel_launch(void* const* d_in, const int* in_sizes, int n_in,
                              void* d_out, int out_size, void* d_ws, size_t ws_size,
                              hipStream_t stream)
{
  (void)in_sizes; (void)out_size; (void)ws_size;
  const float* x      = (const float*)d_in[0];
  const float* eps    = (const float*)d_in[1];
  const float* gum    = (const float*)d_in[2];
  const float* femb_w = (const float*)d_in[3];
  const float* femb_b = (const float*)d_in[4];
  const float* pos    = (const float*)d_in[5];

  static const long leaf_sz[16] = {
    3L*512*512, 3L*512, 3L*512*512, 3L*512, 3L*512*512, 3L*512,
    3L*512*512, 3L*512, 3L*512, 3L*512, 3L*512, 3L*512,
    3L*512*2048, 3L*2048, 3L*2048*512, 3L*512 };

  Blk enc, dec;
  int idx;
  if (n_in >= 50) {                 // pytree-flattened leaves
    for (int i = 0; i < 16; ++i) enc.p[i] = (const float*)d_in[6 + i];
    for (int i = 0; i < 16; ++i) dec.p[i] = (const float*)d_in[22 + i];
    idx = 38;
  } else {                          // tuples passed as concatenated blobs
    const float* eb = (const float*)d_in[6];
    const float* db = (const float*)d_in[7];
    long off = 0;
    for (int i = 0; i < 16; ++i) { enc.p[i] = eb + off; dec.p[i] = db + off; off += leaf_sz[i]; }
    idx = 8;
  }
  const float* fc_mu_W  = (const float*)d_in[idx + 0];
  const float* fc_mu_b  = (const float*)d_in[idx + 1];
  const float* fc_lv_W  = (const float*)d_in[idx + 2];
  const float* fc_lv_b  = (const float*)d_in[idx + 3];
  const float* dec_in_W = (const float*)d_in[idx + 4];
  const float* dec_in_b = (const float*)d_in[idx + 5];
  const float* num_W    = (const float*)d_in[idx + 6];
  const float* num_b    = (const float*)d_in[idx + 7];
  const float* cat_W    = (const float*)d_in[idx + 8];
  const float* cat_b    = (const float*)d_in[idx + 9];
  const int*   num_idx  = (const int*)d_in[idx + 10];
  const int*   cat_idx  = (const int*)d_in[idx + 11];

  // Workspace: 5 x [65536,512] fp32 buffers (128 MB each) + z (512 KB)
  const size_t BUF = (size_t)NTOK * D_MODEL * sizeof(float);
  float* h   = (float*)d_ws;
  float* q   = (float*)((char*)d_ws + 1 * BUF);
  float* kbf = (float*)((char*)d_ws + 2 * BUF);
  float* vbf = (float*)((char*)d_ws + 3 * BUF);
  float* cbf = (float*)((char*)d_ws + 4 * BUF);
  float* z   = (float*)((char*)d_ws + 5 * BUF);

  float* out    = (float*)d_out;
  float* recon  = out;                            // [512,128]
  float* out_mu = out + NTOK;                     // [512,256]
  float* out_lv = out + NTOK + BATCH * LATENT;    // [512,256]

  auto gemm = [&](const float* A, const float* W, const float* bias, float* C,
                  int M, int N, int K, int relu) {
    dim3 g(N / 128, M / 128);
    gemm_wmma_kernel<<<g, 256, 0, stream>>>(A, W, bias, C, M, N, K, relu);
  };

  auto run_stack = [&](const Blk& P) {
    for (int l = 0; l < 3; ++l) {
      const float* Wq  = P.p[0]  + (size_t)l * D_MODEL * D_MODEL;
      const float* bq  = P.p[1]  + (size_t)l * D_MODEL;
      const float* Wk  = P.p[2]  + (size_t)l * D_MODEL * D_MODEL;
      const float* bk  = P.p[3]  + (size_t)l * D_MODEL;
      const float* Wv  = P.p[4]  + (size_t)l * D_MODEL * D_MODEL;
      const float* bv  = P.p[5]  + (size_t)l * D_MODEL;
      const float* Wo  = P.p[6]  + (size_t)l * D_MODEL * D_MODEL;
      const float* bo  = P.p[7]  + (size_t)l * D_MODEL;
      const float* g1  = P.p[8]  + (size_t)l * D_MODEL;
      const float* be1 = P.p[9]  + (size_t)l * D_MODEL;
      const float* g2  = P.p[10] + (size_t)l * D_MODEL;
      const float* be2 = P.p[11] + (size_t)l * D_MODEL;
      const float* W1  = P.p[12] + (size_t)l * D_MODEL * D_FF;
      const float* bf1 = P.p[13] + (size_t)l * D_FF;
      const float* W2  = P.p[14] + (size_t)l * D_FF * D_MODEL;
      const float* bf2 = P.p[15] + (size_t)l * D_MODEL;

      gemm(h, Wq, bq, q,   NTOK, D_MODEL, D_MODEL, 0);
      gemm(h, Wk, bk, kbf, NTOK, D_MODEL, D_MODEL, 0);
      gemm(h, Wv, bv, vbf, NTOK, D_MODEL, D_MODEL, 0);
      attn_kernel<<<dim3(BATCH * NHEAD), 256, 0, stream>>>(q, kbf, vbf, cbf);
      gemm(cbf, Wo, bo, q, NTOK, D_MODEL, D_MODEL, 0);          // q reused
      ln_residual_kernel<<<NTOK, 128, 0, stream>>>(h, q, g1, be1);
      for (int mc = 0; mc < NTOK; mc += FFCHUNK) {              // FFN chunked
        gemm(h + (size_t)mc * D_MODEL, W1, bf1, vbf, FFCHUNK, D_FF, D_MODEL, 1);
        gemm(vbf, W2, bf2, cbf + (size_t)mc * D_MODEL, FFCHUNK, D_MODEL, D_FF, 0);
      }
      ln_residual_kernel<<<NTOK, 128, 0, stream>>>(h, cbf, g2, be2);
    }
  };

  // ---- Encoder ----
  embed_kernel<<<NTOK, 128, 0, stream>>>(x, femb_w, femb_b, pos, h);
  run_stack(enc);

  // ---- Latent ----  (mu/logvar written straight into d_out)
  gemm(h, fc_mu_W, fc_mu_b, out_mu, BATCH, LATENT, SEQ * D_MODEL, 0);
  gemm(h, fc_lv_W, fc_lv_b, out_lv, BATCH, LATENT, SEQ * D_MODEL, 0);
  reparam_kernel<<<(BATCH * LATENT) / 256, 256, 0, stream>>>(out_mu, out_lv, eps, z);
  gemm(z, dec_in_W, dec_in_b, h, BATCH, SEQ * D_MODEL, LATENT, 0);

  // ---- Decoder ----
  run_stack(dec);

  // ---- Heads ----
  num_head_kernel<<<BATCH * 96, 128, 0, stream>>>(h, num_W, num_b, num_idx, recon);
  cat_head_kernel<<<BATCH * NCAT, 256, 0, stream>>>(h, cat_W, cat_b, gum, cat_idx, recon);
}
